// CustomQuantLinear_23252952941035
// MI455X (gfx1250) — matrix operand order repaired and checked by reading, not compiled, not run on hardware
//
#include <hip/hip_runtime.h>

typedef __attribute__((ext_vector_type(16))) _Float16 v16h;
typedef __attribute__((ext_vector_type(8)))  _Float16 v8h;
typedef __attribute__((ext_vector_type(4)))  _Float16 v4h;
typedef __attribute__((ext_vector_type(8)))  float    v8f;
typedef __attribute__((ext_vector_type(4)))  float    v4f;
typedef __attribute__((ext_vector_type(4)))  int      v4i;

#define M_TOT 8192
#define N_TOT 11008
#define K_TOT 4096
#define BM 128
#define BN 128
#define BK 32
#define NK (K_TOT / BK)      // 128 K-steps
#define LDSS 40              // halfs per LDS row (32 data + 8 pad)

__global__ __launch_bounds__(256, 2)
void qlinear_wmma_f16(const float* __restrict__ x,
                      const int* __restrict__ w,
                      const _Float16* __restrict__ scale,
                      const int* __restrict__ zp,
                      const float* __restrict__ bias,
                      float* __restrict__ out)
{
    __shared__ _Float16 As[2][BM * LDSS];   // x tile, f16
    __shared__ _Float16 Bs[2][BN * LDSS];   // (w - zp) tile, f16

    const int tid   = threadIdx.x;
    const int lane  = tid & 31;
    const int ln    = lane & 15;            // n (or m) within fragment
    const int g     = lane >> 4;            // lane half-group
    const int wid   = tid >> 5;             // 0..7
    const int waveM = wid & 1;              // 2 waves along M -> 64 rows each
    const int waveN = wid >> 1;             // 4 waves along N -> 32 cols each

    const int mBase = blockIdx.x * BM;
    const int nBase = blockIdx.y * BN;

    // staging assignment: 8 threads per row (4 elements each), 32 rows/pass, 4 passes
    const int srow = tid >> 3;              // 0..31
    const int skq  = (tid & 7) * 4;         // 0..28

    const float* xg = x + (size_t)mBase * K_TOT;
    const int*   wg = w + (size_t)nBase * K_TOT;

    // zero-points for the 4 weight rows this thread stages (constant over K loop)
    int zpv[4];
#pragma unroll
    for (int p = 0; p < 4; ++p)
        zpv[p] = zp[nBase + srow + 32 * p];

    v8f c[4][2];
#pragma unroll
    for (int i = 0; i < 4; ++i)
#pragma unroll
        for (int j = 0; j < 2; ++j)
            c[i][j] = (v8f){0.f,0.f,0.f,0.f,0.f,0.f,0.f,0.f};

    // in-flight staging registers (loaded early, consumed after compute)
    v4f aReg[4];
    v4i wReg[4];

    auto load = [&](int kt) {
        const int kb = kt * BK;
#pragma unroll
        for (int p = 0; p < 4; ++p) {
            const int row = srow + 32 * p;
            aReg[p] = *(const v4f*)(xg + (size_t)row * K_TOT + kb + skq);
            wReg[p] = *(const v4i*)(wg + (size_t)row * K_TOT + kb + skq);
            if (kt + 2 < NK) {   // gfx1250 speculative prefetch into GL2
                __builtin_prefetch(xg + (size_t)row * K_TOT + kb + 2 * BK + skq, 0, 0);
                __builtin_prefetch(wg + (size_t)row * K_TOT + kb + 2 * BK + skq, 0, 0);
            }
        }
    };

    auto store = [&](int buf) {
#pragma unroll
        for (int p = 0; p < 4; ++p) {
            const int row = srow + 32 * p;
            v4f a4 = aReg[p];
            v4h ah = { (_Float16)a4.x, (_Float16)a4.y,
                       (_Float16)a4.z, (_Float16)a4.w };
            *(v4h*)&As[buf][row * LDSS + skq] = ah;
            v4i w4 = wReg[p];
            v4h bh = { (_Float16)(w4.x - zpv[p]), (_Float16)(w4.y - zpv[p]),
                       (_Float16)(w4.z - zpv[p]), (_Float16)(w4.w - zpv[p]) };
            *(v4h*)&Bs[buf][row * LDSS + skq] = bh;
        }
    };

    auto compute = [&](int buf) {
        const _Float16* Ab = &As[buf][(waveM * 64) * LDSS];
        const _Float16* Bb = &Bs[buf][(waveN * 32) * LDSS];
        // B fragments: lane ln = column n; halves are K = g*16 + 0..15 (contiguous)
        v16h bfrag[2];
#pragma unroll
        for (int j = 0; j < 2; ++j)
            bfrag[j] = *(const v16h*)(Bb + (j * 16 + ln) * LDSS + g * 16);
#pragma unroll
        for (int i = 0; i < 4; ++i) {
            // A fragment: lane ln = row m; halves K = g*16 + 0..15 (contiguous)
            const _Float16* rp = Ab + (i * 16 + ln) * LDSS;
            v8h lo = *(const v8h*)(rp + g * 16);
            v8h hi = *(const v8h*)(rp + g * 16 + 8);
            v16h afrag = __builtin_shufflevector(lo, hi,
                0,1,2,3,4,5,6,7,8,9,10,11,12,13,14,15);
#pragma unroll
            for (int j = 0; j < 2; ++j)
                c[i][j] = __builtin_amdgcn_wmma_f32_16x16x32_f16(
                    false, afrag, false, bfrag[j],
                    (short)0, c[i][j], false, false);
        }
    };

    // prologue: fill buffer 0
    load(0);
    store(0);
    __syncthreads();

#pragma unroll 1
    for (int kt = 0; kt < NK; ++kt) {
        const int cur = kt & 1;
        if (kt + 1 < NK) load(kt + 1);      // issue HBM loads, no wait
        compute(cur);                       // WMMAs hide load latency
        if (kt + 1 < NK) store(cur ^ 1);    // loadcnt wait lands here
        __syncthreads();
    }

    // Epilogue: out = scale[n] * acc + bias[n]
#pragma unroll
    for (int j = 0; j < 2; ++j) {
        const int n = nBase + waveN * 32 + j * 16 + ln;
        const float sc = (float)scale[n];
        const float bs = bias[n];
#pragma unroll
        for (int i = 0; i < 4; ++i) {
            const int m0 = mBase + waveM * 64 + i * 16 + g * 8;
#pragma unroll
            for (int r = 0; r < 8; ++r)
                out[(size_t)(m0 + r) * N_TOT + n] = c[i][j][r] * sc + bs;
        }
    }
}

extern "C" void kernel_launch(void* const* d_in, const int* in_sizes, int n_in,
                              void* d_out, int out_size, void* d_ws, size_t ws_size,
                              hipStream_t stream) {
    const float*    xp  = (const float*)d_in[0];
    const int*      wp  = (const int*)d_in[1];
    const _Float16* sc  = (const _Float16*)d_in[2];
    const int*      zpp = (const int*)d_in[3];
    const float*    bp  = (const float*)d_in[4];
    float*          op  = (float*)d_out;

    dim3 grid(M_TOT / BM, N_TOT / BN);   // 64 x 86; x-fastest keeps one weight
    dim3 block(256);                     // strip hot in L2 across 64 M-tiles
    qlinear_wmma_f16<<<grid, block, 0, stream>>>(xp, wp, sc, zpp, bp, op);
}
